// StaticCacheModel_68186900792253
// MI455X (gfx1250) — matrix operand order repaired and checked by reading, not compiled, not run on hardware
//
#include <hip/hip_runtime.h>
#include <hip/hip_bf16.h>
#include <math.h>

typedef __attribute__((ext_vector_type(16))) _Float16 v16h;
typedef __attribute__((ext_vector_type(8)))  _Float16 v8h;
typedef __attribute__((ext_vector_type(8)))  float    v8f;

#define BATCH   2
#define HEADS   32
#define SLEN    2048
#define DHEAD   128
#define MAXSEQ  8192
#define NBH     (BATCH * HEADS)
#define QKV_N   ((size_t)NBH * SLEN * DHEAD)      // 16,777,216
#define CACHE_N ((size_t)NBH * MAXSEQ * DHEAD)    // 67,108,864

// ---------------------------------------------------------------------------
// Prep 1: assemble full-capacity caches (start_idx=0, end_idx=2048):
// slots [0,2048) come from new K/V, the tail from the old cache.
// Streaming data: non-temporal policies keep L2 for the attention working set.
// ---------------------------------------------------------------------------
__global__ void assemble_cache_kernel(const float* __restrict__ newkv,
                                      const float* __restrict__ cache,
                                      float* __restrict__ outc) {
    size_t idx = (size_t)blockIdx.x * blockDim.x + threadIdx.x;
    if (idx >= CACHE_N) return;
    const size_t perbh = (size_t)MAXSEQ * DHEAD;
    size_t bh  = idx / perbh;
    size_t rem = idx - bh * perbh;
    size_t s   = rem / DHEAD;
    float val;
    if (s < SLEN) {
        val = newkv[bh * (size_t)SLEN * DHEAD + rem];
    } else {
        val = __builtin_nontemporal_load(&cache[idx]);
    }
    __builtin_nontemporal_store(val, &outc[idx]);
}

// ---------------------------------------------------------------------------
// Prep 2: f16 conversions. Q is pre-scaled by 1/sqrt(D).
// ---------------------------------------------------------------------------
__global__ void convert_qk_kernel(const float* __restrict__ q,
                                  const float* __restrict__ k,
                                  _Float16* __restrict__ Qh,
                                  _Float16* __restrict__ Kh,
                                  float scale) {
    size_t idx = (size_t)blockIdx.x * blockDim.x + threadIdx.x;
    if (idx >= QKV_N) return;
    Qh[idx] = (_Float16)(q[idx] * scale);
    Kh[idx] = (_Float16)k[idx];
}

// ---------------------------------------------------------------------------
// Prep 3: V transpose to [bh][d][s] f16 so the PV B-operand is contiguous.
// ---------------------------------------------------------------------------
__global__ void transpose_v_kernel(const float* __restrict__ v,
                                   _Float16* __restrict__ Vt) {
    __shared__ float tile[16][17];
    const int tx = threadIdx.x, ty = threadIdx.y;
    const int d0 = blockIdx.x * 16;
    const int s0 = blockIdx.y * 16;
    const int bh = blockIdx.z;
    tile[ty][tx] = v[((size_t)bh * SLEN + (s0 + ty)) * DHEAD + d0 + tx];
    __syncthreads();
    Vt[((size_t)bh * DHEAD + (d0 + ty)) * SLEN + s0 + tx] = (_Float16)tile[tx][ty];
}

// ---------------------------------------------------------------------------
// DPP16 row_ror rotate within each 16-lane row (VALU pipe, no LDS traffic).
// dpp_ctrl 0x120|N = row_ror:N; row_mask=0xF, bank_mask=0xF, bound_ctrl=1.
// Rotate-halving butterfly (8,4,2,1) all-reduces max/sum over each 16-lane half.
// ---------------------------------------------------------------------------
template <int N>
__device__ __forceinline__ float row_ror(float x) {
    int xi = __builtin_bit_cast(int, x);
    int yi = __builtin_amdgcn_update_dpp(0, xi, 0x120 | N, 0xF, 0xF, true);
    return __builtin_bit_cast(float, yi);
}
__device__ __forceinline__ float rowmax16(float x) {
    x = fmaxf(x, row_ror<8>(x));
    x = fmaxf(x, row_ror<4>(x));
    x = fmaxf(x, row_ror<2>(x));
    x = fmaxf(x, row_ror<1>(x));
    return x;
}
__device__ __forceinline__ float rowsum16(float x) {
    x += row_ror<8>(x);
    x += row_ror<4>(x);
    x += row_ror<2>(x);
    x += row_ror<1>(x);
    return x;
}

// ---------------------------------------------------------------------------
// Flash attention: one wave32 per 16-row Q tile, 32 keys per iteration.
// Main loop is unmasked; exactly one partially-masked tile handled at the end.
// ---------------------------------------------------------------------------
__device__ __forceinline__ v16h load16(const _Float16* p) {
    v8h lo = *(const v8h*)p;
    v8h hi = *(const v8h*)(p + 8);
    v16h r;
#pragma unroll
    for (int e = 0; e < 8; ++e) { r[e] = lo[e]; r[8 + e] = hi[e]; }
    return r;
}

__global__ __launch_bounds__(32) void fa_kernel(const _Float16* __restrict__ Qh,
                                                const _Float16* __restrict__ Kh,
                                                const _Float16* __restrict__ Vt,
                                                float* __restrict__ Out) {
    __shared__ float sP[16][33];   // padded: 33*m mod 64 distinct -> no bank conflicts

    const int lane = threadIdx.x;
    const int n    = lane & 15;    // column within 16-wide tile
    const int half = lane >> 4;    // 0/1 lane-half
    const int tq   = blockIdx.x % (SLEN / 16);
    const int bh   = blockIdx.x / (SLEN / 16);
    const int qbase   = tq * 16;
    const int rowbase = qbase + half * 8;

    // ---- preload Q tile as 4 A-operands (K-chunks of 32 along D) ----
    const _Float16* Qp = Qh + ((size_t)bh * SLEN + (qbase + n)) * DHEAD;
    v16h aQ[4];
#pragma unroll
    for (int c = 0; c < 4; ++c) {
        v8h lo = *(const v8h*)(Qp + c * 32 + half * 16);
        v8h hi = *(const v8h*)(Qp + c * 32 + half * 16 + 8);
#pragma unroll
        for (int e = 0; e < 8; ++e) { aQ[c][e] = lo[e]; aQ[c][8 + e] = hi[e]; }
    }

    // ---- accumulator state ----
    v8f O[8];
#pragma unroll
    for (int t = 0; t < 8; ++t)
#pragma unroll
        for (int r = 0; r < 8; ++r) O[t][r] = 0.0f;
    float mrow[8], lrow[8];
#pragma unroll
    for (int r = 0; r < 8; ++r) { mrow[r] = -__builtin_inff(); lrow[r] = 0.0f; }

    const _Float16* Kbase = Kh + (size_t)bh * SLEN * DHEAD;
    const _Float16* Vbase = Vt + ((size_t)bh * DHEAD + n) * SLEN;

    auto step = [&](int k0, bool masked) __attribute__((always_inline)) {
        // ---- scores: S = Q @ K^T for 2 key sub-tiles of 16 ----
        v8f S0 = {}, S1 = {};
        const _Float16* Kp0 = Kbase + (size_t)(k0 + n) * DHEAD + half * 16;
        const _Float16* Kp1 = Kp0 + (size_t)16 * DHEAD;
#pragma unroll
        for (int c = 0; c < 4; ++c) {
            v16h b0 = load16(Kp0 + c * 32);
            S0 = __builtin_amdgcn_wmma_f32_16x16x32_f16(false, aQ[c], false, b0,
                                                        (short)0, S0, false, false);
            v16h b1 = load16(Kp1 + c * 32);
            S1 = __builtin_amdgcn_wmma_f32_16x16x32_f16(false, aQ[c], false, b1,
                                                        (short)0, S1, false, false);
        }

        // ---- (optional) causal mask + online softmax update ----
        const int col0 = k0 + n;
        const int col1 = col0 + 16;
        float alpha[8];
#pragma unroll
        for (int r = 0; r < 8; ++r) {
            const int row = rowbase + r;
            float s0 = S0[r], s1 = S1[r];
            if (masked) {
                if (col0 > row) s0 = -__builtin_inff();
                if (col1 > row) s1 = -__builtin_inff();
            }
            const float rm   = rowmax16(fmaxf(s0, s1));
            const float mnew = fmaxf(mrow[r], rm);
            alpha[r] = __expf(mrow[r] - mnew);
            const float p0 = __expf(s0 - mnew);
            const float p1 = __expf(s1 - mnew);
            const float rs = rowsum16(p0 + p1);
            lrow[r] = lrow[r] * alpha[r] + rs;
            mrow[r] = mnew;
            sP[half * 8 + r][n]      = p0;
            sP[half * 8 + r][16 + n] = p1;
        }
        __syncthreads();

        // ---- P (C-layout) -> A-layout via LDS ----
        v16h aP;
        const int abase = half * 16;
#pragma unroll
        for (int e = 0; e < 8; ++e) {
            aP[e]     = (_Float16)sP[n][abase + e];
            aP[8 + e] = (_Float16)sP[n][abase + 8 + e];
        }
        __syncthreads();

        // ---- rescale O, then O += P @ V ----
#pragma unroll
        for (int t = 0; t < 8; ++t)
#pragma unroll
            for (int r = 0; r < 8; ++r) O[t][r] *= alpha[r];

        const _Float16* Vp = Vbase + k0 + half * 16;
#pragma unroll
        for (int t = 0; t < 8; ++t) {
            v16h bV = load16(Vp + (size_t)t * 16 * SLEN);
            O[t] = __builtin_amdgcn_wmma_f32_16x16x32_f16(false, aP, false, bV,
                                                          (short)0, O[t], false, false);
        }
    };

    // Tiles with k0+31 < qbase are never masked (min row = qbase).
    const int kmask = qbase & ~31;           // the single partially-masked tile
    for (int k0 = 0; k0 < kmask; k0 += 32)
        step(k0, false);
    step(kmask, true);

    // ---- normalize and write out (streaming stores) ----
    float inv[8];
#pragma unroll
    for (int r = 0; r < 8; ++r) inv[r] = 1.0f / lrow[r];
    float* Op = Out + (size_t)bh * SLEN * DHEAD;
#pragma unroll
    for (int t = 0; t < 8; ++t)
#pragma unroll
        for (int r = 0; r < 8; ++r)
            __builtin_nontemporal_store(O[t][r] * inv[r],
                &Op[(size_t)(rowbase + r) * DHEAD + 16 * t + n]);
}

// ---------------------------------------------------------------------------
extern "C" void kernel_launch(void* const* d_in, const int* in_sizes, int n_in,
                              void* d_out, int out_size, void* d_ws, size_t ws_size,
                              hipStream_t stream) {
    const float* q  = (const float*)d_in[0];
    const float* k  = (const float*)d_in[1];
    const float* v  = (const float*)d_in[2];
    const float* kc = (const float*)d_in[3];
    const float* vc = (const float*)d_in[4];

    float* out = (float*)d_out;                 // [B,H,S,D]
    float* okc = out + QKV_N;                   // [B,H,MAX,D]
    float* ovc = okc + CACHE_N;                 // [B,H,MAX,D]

    _Float16* Qh = (_Float16*)d_ws;             // f16, pre-scaled by 1/sqrt(D)
    _Float16* Kh = Qh + QKV_N;                  // f16
    _Float16* Vt = Kh + QKV_N;                  // f16, transposed [bh][d][s]

    const float scale = 0.08838834764831845f;   // 1/sqrt(128)

    // cache assembly (the two extra outputs)
    {
        const int threads = 256;
        const int blocks  = (int)((CACHE_N + threads - 1) / threads);
        assemble_cache_kernel<<<blocks, threads, 0, stream>>>(k, kc, okc);
        assemble_cache_kernel<<<blocks, threads, 0, stream>>>(v, vc, ovc);
    }
    // f16 prep
    {
        const int threads = 256;
        const int blocks  = (int)((QKV_N + threads - 1) / threads);
        convert_qk_kernel<<<blocks, threads, 0, stream>>>(q, k, Qh, Kh, scale);
    }
    {
        dim3 grid(DHEAD / 16, SLEN / 16, NBH);
        dim3 block(16, 16);
        transpose_v_kernel<<<grid, block, 0, stream>>>(v, Vt);
    }
    // flash attention: one wave per 16-row query tile
    {
        const int blocks = NBH * (SLEN / 16);   // 8192 waves
        fa_kernel<<<blocks, 32, 0, stream>>>(Qh, Kh, Vt, out);
    }
}
